// EncoderLSTM_60370060312745
// MI455X (gfx1250) — matrix operand order repaired and checked by reading, not compiled, not run on hardware
//
#include <hip/hip_runtime.h>

// ---------------------------------------------------------------------------
// 2-layer LSTM encoder for MI455X (gfx1250), wave32 + WMMA bf16.
//   V=32000, E=512, H=1024, N=16, L=128
// Strategy:
//   1. Pack all weights fp32->bf16 into WMMA B-tile register layout (L2-resident,
//      28 MB total << 192 MB L2).
//   2. Embedding gather + layer-0 input GEMM for ALL timesteps as one batched
//      WMMA GEMM (parallel over 128*256 tiles), bias fused, stored in C-layout.
//   3. Sequential recurrence, software-pipelined across layers: launch k runs
//      layer0(t=k) on waves 0..63 and layer1(t=k-1) on waves 64..127 (they are
//      independent), so the chain needs 129 launches instead of 256. Each wave
//      owns one 16-wide hidden slice and computes all 4 gates locally.
// ---------------------------------------------------------------------------

typedef __attribute__((ext_vector_type(16))) __bf16 v16bf;
typedef __attribute__((ext_vector_type(8)))  float  v8f;

#define SEQ    128
#define HID    1024
#define EMBD   512
#define NTILES 256   // 4H / 16
#define JTILES 64    // H / 16

__device__ __forceinline__ float sigmoidf_(float x) {
    return 1.0f / (1.0f + __expf(-x));
}
__device__ __forceinline__ float tanhf_(float x) {
    x = fminf(fmaxf(x, -20.0f), 20.0f);
    float e = __expf(2.0f * x);
    return (e - 1.0f) / (e + 1.0f);
}

// --- Pack a [4H, Kdim] fp32 weight into WMMA B-tile bf16 layout -------------
// Tile (nt, kt) covers rows nt*16..+15 (output cols N) and K kt*32..+31.
// B 32x16 layout: lane<16 -> N=lane,    elems 0..7 = K 0..7,  8..15 = K 16..23
//                 lane>=16 -> N=lane-16, elems 0..7 = K 8..15, 8..15 = K 24..31
__global__ void pack_w_kernel(const float* __restrict__ w, __bf16* __restrict__ out,
                              int nK, int Kdim) {
    int gid  = blockIdx.x * blockDim.x + threadIdx.x;
    int total = NTILES * nK * 32;
    if (gid >= total) return;
    int lane = gid & 31;
    int tile = gid >> 5;
    int nt = tile / nK, kt = tile - nt * nK;
    int n = nt * 16 + (lane & 15);
    int kbase = kt * 32 + ((lane & 16) ? 8 : 0);
    __bf16* o = out + (size_t)tile * 512 + lane * 16;
    const float* src = w + (size_t)n * Kdim + kbase;
#pragma unroll
    for (int j = 0; j < 16; ++j) o[j] = (__bf16)src[j < 8 ? j : j + 8];
}

// --- Embedding gather -> WMMA A-tile bf16 layout ----------------------------
// A 16x32 layout: lane<16 -> M=lane, elems 0..7 = K 0..7, elems 8..15 = K 16..23
//                 lane>=16 -> M=lane-16, elems 0..7 = K 8..15, 8..15 = K 24..31
__global__ void pack_x0_kernel(const int* __restrict__ x, const float* __restrict__ emb,
                               __bf16* __restrict__ out) {
    int gid = blockIdx.x * blockDim.x + threadIdx.x;  // SEQ*16*32
    if (gid >= SEQ * 16 * 32) return;
    int lane = gid & 31;
    int tk   = gid >> 5;     // t*16 + kt
    int kt   = tk & 15;
    int t    = tk >> 4;
    int row  = lane & 15;
    int tok  = x[row * SEQ + t];
    const float* e = emb + (size_t)tok * EMBD + kt * 32;
    int base = (lane & 16) ? 8 : 0;
    __bf16* o = out + (size_t)tk * 512 + lane * 16;
#pragma unroll
    for (int j = 0; j < 8; ++j) o[j]     = (__bf16)e[base + j];
#pragma unroll
    for (int j = 0; j < 8; ++j) o[8 + j] = (__bf16)e[base + 16 + j];
}

// --- Batched layer-0 input projection: G0in[t] = X0[t] @ w_ih0^T + b0 -------
// One wave per (t, ntile); K = 512 -> 16 WMMA steps. Output in C-layout
// ([lane][8] fp32 per tile) so the recurrence initializes accumulators directly.
__global__ void gemm_in0_kernel(const __bf16* __restrict__ X0p, const __bf16* __restrict__ Wp,
                                const float* __restrict__ bias, float* __restrict__ G0in) {
    int wid  = blockIdx.x * (blockDim.x >> 5) + (threadIdx.x >> 5);
    int lane = threadIdx.x & 31;
    if (wid >= SEQ * NTILES) return;
    int t     = wid >> 8;
    int ntile = wid & 255;
    v8f acc = {};
    const __bf16* Abase = X0p + (size_t)t * 16 * 512 + lane * 16;
    const __bf16* Bbase = Wp  + (size_t)ntile * 16 * 512 + lane * 16;
#pragma unroll 4
    for (int kt = 0; kt < 16; ++kt) {
        v16bf a = *(const v16bf*)(Abase + kt * 512);
        v16bf b = *(const v16bf*)(Bbase + kt * 512);
        acc = __builtin_amdgcn_wmma_f32_16x16x32_bf16(false, a, false, b,
                                                      (short)0, acc, false, false);
    }
    float bv = bias[ntile * 16 + (lane & 15)];
#pragma unroll
    for (int r = 0; r < 8; ++r) acc[r] += bv;
    *(v8f*)(G0in + (size_t)wid * 256 + lane * 8) = acc;
}

// --- LSTM cell nonlinearity + scatter of new h into A-tile layout -----------
// C/D layout: element r of lane l -> (row = l<16 ? r : r+8, col = jt*16 + l%16)
__device__ __forceinline__ void cell_update_scatter(
    v8f acc[4], float* __restrict__ cp, __bf16* __restrict__ hpackW,
    int jt, int lane, float* __restrict__ out_t) {
#pragma unroll
    for (int r = 0; r < 8; ++r) {
        float iv = sigmoidf_(acc[0][r]);
        float fv = sigmoidf_(acc[1][r]);
        float gv = tanhf_(acc[2][r]);
        float ov = sigmoidf_(acc[3][r]);
        float c  = fv * cp[r] + iv * gv;
        cp[r] = c;
        float h = ov * tanhf_(c);
        int row = (lane < 16) ? r : r + 8;
        int col = jt * 16 + (lane & 15);
        // scatter h into A-tile layout for the next GEMM
        int kt2 = col >> 5;
        int kl  = col & 31;
        int lp, j;
        if      (kl < 8)  { lp = row;      j = kl;      }
        else if (kl < 16) { lp = row + 16; j = kl - 8;  }
        else if (kl < 24) { lp = row;      j = kl - 8;  }
        else              { lp = row + 16; j = kl - 16; }
        hpackW[(size_t)kt2 * 512 + lp * 16 + j] = (__bf16)h;
        if (out_t) out_t[(size_t)row * (SEQ * HID) + col] = h;
    }
}

// --- Fused recurrent step (software-pipelined across layers) ----------------
// Waves  0..63  : layer0 at t=k   : gates = G0in[k] + h0(k-1) @ w_hh0^T
// Waves 64..127 : layer1 at t=k-1 : gates = h0(k-1) @ w_ih1^T + h1(k-2) @ w_hh1^T + b1
// The two halves touch disjoint buffers (layer1 only READS h0R, which layer0
// also only reads), so no intra-launch ordering is required.
__global__ void step_fused_kernel(
    const __bf16* __restrict__ h0R, __bf16* __restrict__ h0W,
    const __bf16* __restrict__ Whh0,
    const float* __restrict__ G0in_t, float* __restrict__ c0buf,
    const __bf16* __restrict__ h1R, __bf16* __restrict__ h1W,
    const __bf16* __restrict__ Wih1, const __bf16* __restrict__ Whh1,
    const float* __restrict__ b1, float* __restrict__ c1buf,
    float* __restrict__ out_t, int do_l0, int do_l1) {
    int wid  = blockIdx.x * (blockDim.x >> 5) + (threadIdx.x >> 5);
    int lane = threadIdx.x & 31;

    if (wid < JTILES) {
        // ----------------- layer 0, t = k -----------------
        if (!do_l0) return;
        int jt = wid;
        v8f acc[4];
#pragma unroll
        for (int g = 0; g < 4; ++g)
            acc[g] = *(const v8f*)(G0in_t + (size_t)(g * JTILES + jt) * 256 + lane * 8);
        const __bf16* Ab = h0R + lane * 16;
#pragma unroll 2
        for (int kt = 0; kt < 32; ++kt) {
            v16bf a = *(const v16bf*)(Ab + kt * 512);
#pragma unroll
            for (int g = 0; g < 4; ++g) {
                const __bf16* bp = Whh0 + ((size_t)(g * JTILES + jt) * 32 + kt) * 512 + lane * 16;
                __builtin_prefetch(bp + 512, 0, 1);
                v16bf b = *(const v16bf*)bp;
                acc[g] = __builtin_amdgcn_wmma_f32_16x16x32_bf16(false, a, false, b,
                                                                 (short)0, acc[g], false, false);
            }
        }
        float* cp = c0buf + (size_t)(jt * 32 + lane) * 8;
        cell_update_scatter(acc, cp, h0W, jt, lane, nullptr);
    } else {
        // ----------------- layer 1, t = k-1 -----------------
        if (!do_l1) return;
        int jt = wid - JTILES;
        v8f acc[4];
#pragma unroll
        for (int g = 0; g < 4; ++g) {
            float bv = b1[g * HID + jt * 16 + (lane & 15)];
#pragma unroll
            for (int r = 0; r < 8; ++r) acc[g][r] = bv;
        }
        const __bf16* A0 = h0R + lane * 16;   // h0(k-1), produced last launch
#pragma unroll 2
        for (int kt = 0; kt < 32; ++kt) {
            v16bf a = *(const v16bf*)(A0 + kt * 512);
#pragma unroll
            for (int g = 0; g < 4; ++g) {
                const __bf16* bp = Wih1 + ((size_t)(g * JTILES + jt) * 32 + kt) * 512 + lane * 16;
                __builtin_prefetch(bp + 512, 0, 1);
                v16bf b = *(const v16bf*)bp;
                acc[g] = __builtin_amdgcn_wmma_f32_16x16x32_bf16(false, a, false, b,
                                                                 (short)0, acc[g], false, false);
            }
        }
        const __bf16* A1 = h1R + lane * 16;   // h1(k-2)
#pragma unroll 2
        for (int kt = 0; kt < 32; ++kt) {
            v16bf a = *(const v16bf*)(A1 + kt * 512);
#pragma unroll
            for (int g = 0; g < 4; ++g) {
                const __bf16* bp = Whh1 + ((size_t)(g * JTILES + jt) * 32 + kt) * 512 + lane * 16;
                __builtin_prefetch(bp + 512, 0, 1);
                v16bf b = *(const v16bf*)bp;
                acc[g] = __builtin_amdgcn_wmma_f32_16x16x32_bf16(false, a, false, b,
                                                                 (short)0, acc[g], false, false);
            }
        }
        float* cp = c1buf + (size_t)(jt * 32 + lane) * 8;
        cell_update_scatter(acc, cp, h1W, jt, lane, out_t);
    }
}

// ---------------------------------------------------------------------------
extern "C" void kernel_launch(void* const* d_in, const int* in_sizes, int n_in,
                              void* d_out, int out_size, void* d_ws, size_t ws_size,
                              hipStream_t stream) {
    const int*   x    = (const int*)d_in[0];
    const float* emb  = (const float*)d_in[1];
    const float* wih0 = (const float*)d_in[2];
    const float* whh0 = (const float*)d_in[3];
    const float* b0   = (const float*)d_in[4];
    const float* wih1 = (const float*)d_in[5];
    const float* whh1 = (const float*)d_in[6];
    const float* b1   = (const float*)d_in[7];
    float* out = (float*)d_out;

    // workspace partition (all offsets 1 KB aligned); total 65,273,856 B
    char* ws = (char*)d_ws;
    __bf16* WIH0P = (__bf16*)(ws + 0);            //  4 MB: 256x16 tiles
    __bf16* WHH0P = (__bf16*)(ws + 4194304);      //  8 MB: 256x32 tiles
    __bf16* WIH1P = (__bf16*)(ws + 12582912);     //  8 MB
    __bf16* WHH1P = (__bf16*)(ws + 20971520);     //  8 MB
    __bf16* X0P   = (__bf16*)(ws + 29360128);     //  2 MB: 128x16 A-tiles
    float*  G0IN  = (float*) (ws + 31457280);     // 32 MB: 128x256 C-tiles
    char*   HC    = ws + 65011712;                // state: 256 KB
    __bf16* H0P[2] = { (__bf16*)(HC),           (__bf16*)(HC + 32768) };
    __bf16* H1P[2] = { (__bf16*)(HC + 65536),   (__bf16*)(HC + 98304) };
    float*  C0 = (float*)(HC + 131072);
    float*  C1 = (float*)(HC + 196608);

    // zero h/c state every call (deterministic; ws is poisoned)
    hipMemsetAsync(HC, 0, 262144, stream);

    // one-time packing (cheap, fully parallel)
    pack_w_kernel<<<(NTILES * 16 * 32) / 256, 256, 0, stream>>>(wih0, WIH0P, 16, EMBD);
    pack_w_kernel<<<(NTILES * 32 * 32) / 256, 256, 0, stream>>>(whh0, WHH0P, 32, HID);
    pack_w_kernel<<<(NTILES * 32 * 32) / 256, 256, 0, stream>>>(wih1, WIH1P, 32, HID);
    pack_w_kernel<<<(NTILES * 32 * 32) / 256, 256, 0, stream>>>(whh1, WHH1P, 32, HID);
    pack_x0_kernel<<<(SEQ * 16 * 32) / 256, 256, 0, stream>>>(x, emb, X0P);

    // batched layer-0 input projection over all timesteps (32768 waves)
    gemm_in0_kernel<<<(SEQ * NTILES) / 8, 256, 0, stream>>>(X0P, WIH0P, b0, G0IN);

    // pipelined recurrence: launch k = layer0(t=k) + layer1(t=k-1), 129 launches
    for (int k = 0; k <= SEQ; ++k) {
        int rb = k & 1, wb = rb ^ 1;
        int do_l0 = (k < SEQ);
        int do_l1 = (k > 0);
        const float* g0t = G0IN + (size_t)(do_l0 ? k : 0) * NTILES * 256;
        float* ot = out + (size_t)(do_l1 ? k - 1 : 0) * HID;
        step_fused_kernel<<<16, 256, 0, stream>>>(
            H0P[rb], H0P[wb], WHH0P, g0t, C0,
            H1P[wb], H1P[rb], WIH1P, WHH1P, b1, C1,
            ot, do_l0, do_l1);
    }
}
